// HyperbolicGCNConv_50792283242937
// MI455X (gfx1250) — compile-verified, matrix operand order checked by reading, not run on hardware
//
#include <hip/hip_runtime.h>
#include <math.h>

// ---------------------------------------------------------------------------
// HyperbolicGCNConv for MI455X (gfx1250, wave32)
//   x:[N,128] f32, edge_index:[2,E] i32, W:[128,128] f32, b:[128] f32
//   out = expmap0( segsum_dst( (logmap0(x) @ W^T + b)[src] * dinv[src]*dinv[dst] ) )
// GEMM runs on V_WMMA_F32_16X16X4_F32 (fp32 matrix path). Everything else is
// L2-resident gather/scatter (h + out ~ 51MB << 192MB global L2).
// ---------------------------------------------------------------------------

#define D_FEAT 128

typedef __attribute__((ext_vector_type(2)))  float  v2f;
typedef __attribute__((ext_vector_type(8)))  float  v8f;
typedef __attribute__((ext_vector_type(16))) __bf16 v16bf;

#if defined(__has_builtin)
#  if __has_builtin(__builtin_amdgcn_wmma_f32_16x16x4_f32)
#    define USE_F32_WMMA 1
#  else
#    define USE_F32_WMMA 0
#  endif
#else
#  define USE_F32_WMMA 0
#endif

__device__ __forceinline__ float wave_sum32(float v) {
  #pragma unroll
  for (int off = 16; off > 0; off >>= 1)
    v += __shfl_xor(v, off, 32);
  return v;
}

// ---------------- kernel: zero a float buffer --------------------------------
__global__ void hgcn_fill0(float* __restrict__ p, long n) {
  long i = (long)blockIdx.x * blockDim.x + threadIdx.x;
  if (i < n) p[i] = 0.0f;
}

// ---------------- kernel: per-row logmap0 scale ------------------------------
// scale[n] = artanh(clip(||x_n||, eps, 1-1e-5)) / max(||x_n||, eps)
__global__ void hgcn_logmap_scale(const float* __restrict__ x,
                                  float* __restrict__ rowscale, int N) {
  int wave = (int)((blockIdx.x * (long)blockDim.x + threadIdx.x) >> 5);
  int lane = threadIdx.x & 31;
  if (wave >= N) return;
  const float4 v = ((const float4*)(x + (size_t)wave * D_FEAT))[lane]; // 32*4 = 128
  float ss = v.x * v.x + v.y * v.y + v.z * v.z + v.w * v.w;
  ss = wave_sum32(ss);
  const float norm = sqrtf(ss);
  const float nc   = fminf(fmaxf(norm, 1e-15f), 1.0f - 1e-5f);
  const float at   = 0.5f * logf((1.0f + nc) / (1.0f - nc)); // artanh
  const float sc   = at / fmaxf(norm, 1e-15f);
  if (lane == 0) rowscale[wave] = sc;
}

// ---------------- kernel: symmetric degree -----------------------------------
__global__ void hgcn_degree(const int* __restrict__ src, const int* __restrict__ dst,
                            float* __restrict__ deg, int E) {
  int e = blockIdx.x * blockDim.x + threadIdx.x;
  if (e >= E) return;
  atomicAdd(&deg[src[e]], 1.0f);
  atomicAdd(&deg[dst[e]], 1.0f);
}

__global__ void hgcn_deg_to_dinv(float* __restrict__ deg, int N) {
  int i = blockIdx.x * blockDim.x + threadIdx.x;
  if (i < N) deg[i] = rsqrtf(fmaxf(deg[i], 1.0f));
}

// ---------------- kernel: h = (rowscale * x) @ W^T + b via WMMA --------------
// One block = 256 threads = 8 waves covers a 16-row strip; wave w owns the
// 16-wide output-column tile [16w, 16w+16).  A is (16 x K) f32, B is (K x 16)
// f32 with B[k][n] = W[n][k] (row-major W => contiguous K per lane).
// fp32 WMMA lane layout (ISA 7.12.2): lanes 0-15 <-> half=0 (K pair k0,k0+1),
// lanes 16-31 <-> half=1 (K pair k0+2,k0+3); C/D element i at lane l is
// row (i + 8*half), col l of the tile.
__global__ void __launch_bounds__(256)
hgcn_tangent_gemm(const float* __restrict__ x, const float* __restrict__ rowscale,
                  const float* __restrict__ W, const float* __restrict__ bias,
                  float* __restrict__ h, int N) {
  const int Mbase = blockIdx.x * 16;
  const int wv    = threadIdx.x >> 5;
  const int lane  = threadIdx.x & 31;
  const int half  = lane >> 4;
  const int l     = lane & 15;
  const int Nbase = wv * 16;

  if (Mbase + 16 <= N) {                      // full tile: WMMA (uniform branch)
    const int arow = Mbase + l;               // A row held by this lane
    const int bcol = Nbase + l;               // B column held by this lane
    const float rs = rowscale[arow];
    const float* __restrict__ xr = x + (size_t)arow * D_FEAT;
    const float* __restrict__ wr = W + (size_t)bcol * D_FEAT;
    v8f acc = {};
#if USE_F32_WMMA
    #pragma unroll 4
    for (int k0 = 0; k0 < D_FEAT; k0 += 4) {
      const int ka = k0 + 2 * half;           // this lane's K pair
      const float2 av = *(const float2*)(xr + ka);
      const float2 bv = *(const float2*)(wr + ka);
      v2f a; a.x = av.x * rs; a.y = av.y * rs;
      v2f b; b.x = bv.x;      b.y = bv.y;
      acc = __builtin_amdgcn_wmma_f32_16x16x4_f32(false, a, false, b,
                                                  (short)0, acc, false, false);
    }
#else
    // fallback: bf16 inputs, f32 accumulate (codegen-confirmed builtin).
    // 16-bit A 16x32 layout: VGPR v (v<4): K = 2v + 8*half (+1); VGPR v+4: +16.
    for (int k0 = 0; k0 < D_FEAT; k0 += 32) {
      v16bf a, b;
      #pragma unroll
      for (int v = 0; v < 4; ++v) {
        const int k = k0 + 2 * v + 8 * half;
        a[2*v]     = (__bf16)(xr[k]      * rs);
        a[2*v + 1] = (__bf16)(xr[k + 1]  * rs);
        a[2*v + 8] = (__bf16)(xr[k + 16] * rs);
        a[2*v + 9] = (__bf16)(xr[k + 17] * rs);
        b[2*v]     = (__bf16)wr[k];
        b[2*v + 1] = (__bf16)wr[k + 1];
        b[2*v + 8] = (__bf16)wr[k + 16];
        b[2*v + 9] = (__bf16)wr[k + 17];
      }
      acc = __builtin_amdgcn_wmma_f32_16x16x32_bf16(false, a, false, b,
                                                    (short)0, acc, false, false);
    }
#endif
    const float bb = bias[Nbase + l];
    #pragma unroll
    for (int i = 0; i < 8; ++i)
      h[(size_t)(Mbase + i + 8 * half) * D_FEAT + (Nbase + l)] = acc[i] + bb;
  } else {                                    // ragged tail strip: scalar path
    for (int r = Mbase; r < N; ++r) {
      const float rs = rowscale[r];
      const float* __restrict__ xr = x + (size_t)r * D_FEAT;
      for (int c = (int)threadIdx.x; c < D_FEAT; c += (int)blockDim.x) {
        const float* __restrict__ wr = W + (size_t)c * D_FEAT;
        float acc = 0.0f;
        for (int k = 0; k < D_FEAT; ++k) acc = fmaf(xr[k] * rs, wr[k], acc);
        h[(size_t)r * D_FEAT + c] = acc + bias[c];
      }
    }
  }
}

// ---------------- kernel: edge gather/scale/scatter --------------------------
// One wave per edge; lane handles a float4 chunk (32 lanes * 16B = 512B row).
// h and out live in the 192MB L2, so the random traffic stays on-chip.
__global__ void hgcn_edge_scatter(const int* __restrict__ src, const int* __restrict__ dst,
                                  const float* __restrict__ dinv,
                                  const float* __restrict__ h,
                                  float* __restrict__ out, int E) {
  int wave = (int)((blockIdx.x * (long)blockDim.x + threadIdx.x) >> 5);
  int lane = threadIdx.x & 31;
  if (wave >= E) return;
  const int s = src[wave];
  const int d = dst[wave];
  const float w = dinv[s] * dinv[d];
  const float4 v = ((const float4*)(h + (size_t)s * D_FEAT))[lane];
  float* o = out + (size_t)d * D_FEAT + lane * 4;
  atomicAdd(o + 0, v.x * w);
  atomicAdd(o + 1, v.y * w);
  atomicAdd(o + 2, v.z * w);
  atomicAdd(o + 3, v.w * w);
}

// ---------------- kernel: in-place expmap0 on the accumulator ----------------
__global__ void hgcn_expmap_inplace(float* __restrict__ out, int N) {
  int wave = (int)((blockIdx.x * (long)blockDim.x + threadIdx.x) >> 5);
  int lane = threadIdx.x & 31;
  if (wave >= N) return;
  float4* row = (float4*)(out + (size_t)wave * D_FEAT);
  float4 v = row[lane];
  float ss = v.x * v.x + v.y * v.y + v.z * v.z + v.w * v.w;
  ss = wave_sum32(ss);                        // every lane gets the sum
  const float norm = sqrtf(ss);
  const float sc   = tanhf(norm) / fmaxf(norm, 1e-15f);
  v.x *= sc; v.y *= sc; v.z *= sc; v.w *= sc;
  row[lane] = v;
}

// ---------------------------------------------------------------------------
extern "C" void kernel_launch(void* const* d_in, const int* in_sizes, int n_in,
                              void* d_out, int out_size, void* d_ws, size_t ws_size,
                              hipStream_t stream) {
  const float* x    = (const float*)d_in[0];
  const int*   ei   = (const int*)d_in[1];
  const float* W    = (const float*)d_in[2];
  const float* bias = (const float*)d_in[3];
  float*       out  = (float*)d_out;

  const int N = in_sizes[0] / D_FEAT;
  const int E = in_sizes[1] / 2;
  const int* src = ei;
  const int* dst = ei + E;

  // workspace layout: rowscale[N] | dinv[N] | h[N*128]
  float* rowscale = (float*)d_ws;
  float* dinv     = rowscale + N;
  float* h        = dinv + N;

  const int TB = 256;

  // zero accumulators (harness poisons buffers; we must init every call)
  {
    long nd = (long)N;
    hgcn_fill0<<<(int)((nd + TB - 1) / TB), TB, 0, stream>>>(dinv, nd);
    long no = (long)N * D_FEAT;
    hgcn_fill0<<<(int)((no + TB - 1) / TB), TB, 0, stream>>>(out, no);
  }

  // logmap0 row scales: one wave32 per node
  {
    long threads = (long)N * 32;
    hgcn_logmap_scale<<<(int)((threads + TB - 1) / TB), TB, 0, stream>>>(x, rowscale, N);
  }

  // symmetric degree -> dinv = rsqrt(max(deg,1))
  hgcn_degree<<<(E + TB - 1) / TB, TB, 0, stream>>>(src, dst, dinv, E);
  hgcn_deg_to_dinv<<<(N + TB - 1) / TB, TB, 0, stream>>>(dinv, N);

  // WMMA GEMM: 16-row strip per block, 8 waves x 16-col tiles
  hgcn_tangent_gemm<<<(N + 15) / 16, TB, 0, stream>>>(x, rowscale, W, bias, h, N);

  // gather-scale-scatter: one wave per edge
  {
    long threads = (long)E * 32;
    hgcn_edge_scatter<<<(int)((threads + TB - 1) / TB), TB, 0, stream>>>(src, dst, dinv, h, out, E);
  }

  // expmap0 in place on the accumulated output
  {
    long threads = (long)N * 32;
    hgcn_expmap_inplace<<<(int)((threads + TB - 1) / TB), TB, 0, stream>>>(out, N);
  }
}
